// Attention_8624294330458
// MI455X (gfx1250) — matrix operand order repaired and checked by reading, not compiled, not run on hardware
//
#include <hip/hip_runtime.h>

// ---------------------------------------------------------------------------
// Problem constants (match reference)
// ---------------------------------------------------------------------------
#define NB      4        // batch
#define NS      2048     // sequence
#define NE      1024     // embedding = HEADS*HEAD_DIM
#define NH      16       // heads
#define HD      64       // head dim
#define NROWS   (NB*NS)  // 8192
#define SCALE   0.125f
#define MASKPEN (-50.0f)

typedef unsigned short u16;
typedef unsigned int   u32;

typedef __attribute__((ext_vector_type(16))) __bf16 v16bf;
typedef __attribute__((ext_vector_type(8)))  float  v8f;

union ABFrag {
    v16bf v;
    u32   u[8];
    u16   s[16];
};

__device__ __forceinline__ u16 f32_to_bf16(float f) {
    u32 u = __float_as_uint(f);
    u32 r = u + 0x7FFFu + ((u >> 16) & 1u);   // round-to-nearest-even
    return (u16)(r >> 16);
}

__device__ __forceinline__ v8f wmma_bf16(const ABFrag& a, const ABFrag& b, v8f c) {
    return __builtin_amdgcn_wmma_f32_16x16x32_bf16(
        /*neg_a=*/false, a.v, /*neg_b=*/false, b.v,
        /*c_mod=*/(short)0, c, /*reuse_a=*/false, /*reuse_b=*/false);
}

// ---------------------------------------------------------------------------
// Tensor Data Mover support (gfx1250). Guarded: fallback = synchronous stage.
// ---------------------------------------------------------------------------
#if defined(__AMDGCN__) && __has_builtin(__builtin_amdgcn_tensor_load_to_lds) && \
    __has_builtin(__builtin_amdgcn_s_wait_tensorcnt)
#define HAVE_TDM 1
#else
#define HAVE_TDM 0
#endif

#if HAVE_TDM
typedef __attribute__((ext_vector_type(4))) unsigned int tdm_v4u;
typedef __attribute__((ext_vector_type(8))) int          tdm_v8i;
typedef __attribute__((ext_vector_type(4))) int          tdm_v4i;

// 2D tile DMA: global (row-major, 2-byte elems) -> LDS.
// Descriptor layout per CDNA5 ISA.
__device__ __forceinline__ void tdm_load_tile_2d(unsigned long long gaddr_bytes,
                                                 unsigned lds_bytes,
                                                 unsigned tile_rows,
                                                 unsigned tile_cols,      // elems
                                                 unsigned tensor_cols,    // dim0
                                                 unsigned tensor_rows,    // dim1
                                                 unsigned long long row_stride_elems) {
    tdm_v4u g0;
    g0.x = 1u;                                                   // count=1 (valid)
    g0.y = lds_bytes;                                            // lds_addr
    g0.z = (unsigned)(gaddr_bytes & 0xFFFFFFFFull);              // global_addr lo
    g0.w = (unsigned)((gaddr_bytes >> 32) & 0x1FFFFFFull)        // global_addr hi
         | (2u << 30);                                           // type = 2 ("image")

    unsigned long long s1 = row_stride_elems * (unsigned long long)tensor_rows;
    tdm_v8i g1;
    g1[0] = (int)(1u << 16);                                     // mask=0, data_size=1 (2B)
    g1[1] = (int)((tensor_cols & 0xFFFFu) << 16);                // tensor_dim0[15:0]
    g1[2] = (int)((tensor_cols >> 16) | ((tensor_rows & 0xFFFFu) << 16));
    g1[3] = (int)((tensor_rows >> 16) | ((tile_cols & 0xFFFFu) << 16));  // tile_dim0
    g1[4] = (int)(tile_rows & 0xFFFFu);                          // tile_dim1, tile_dim2=0
    g1[5] = (int)(unsigned)(row_stride_elems & 0xFFFFFFFFull);   // dim0_stride[31:0]
    g1[6] = (int)((unsigned)((row_stride_elems >> 32) & 0xFFFFull)
                  | ((unsigned)(s1 & 0xFFFFull) << 16));         // dim0_stride hi | dim1_stride lo
    g1[7] = (int)(unsigned)((s1 >> 16) & 0xFFFFFFFFull);         // dim1_stride[47:16]

    tdm_v4i gz = {0, 0, 0, 0};
#if __clang_major__ >= 23
    tdm_v8i gz8 = {0, 0, 0, 0, 0, 0, 0, 0};
    __builtin_amdgcn_tensor_load_to_lds(g0, g1, gz, gz, gz8, 0);
#else
    __builtin_amdgcn_tensor_load_to_lds(g0, g1, gz, gz, 0);
#endif
}
#endif  // HAVE_TDM

// ---------------------------------------------------------------------------
// Kernel 1: elementwise f32 -> bf16
// ---------------------------------------------------------------------------
__global__ void cvt_bf16_kernel(const float* __restrict__ src,
                                u16* __restrict__ dst, int n) {
    int i = blockIdx.x * blockDim.x + threadIdx.x;
    if (i < n) dst[i] = f32_to_bf16(src[i]);
}

// ---------------------------------------------------------------------------
// Kernel 2: weight [K=1024][N=1024] f32 -> transposed bf16 [N][K]
// ---------------------------------------------------------------------------
__global__ void cvt_transpose_kernel(const float* __restrict__ src,
                                     u16* __restrict__ dst) {
    int n     = blockIdx.x * 32 + (threadIdx.x & 31);
    int kbase = blockIdx.y * 32;
#pragma unroll
    for (int kk = 0; kk < 32; ++kk) {
        dst[(size_t)n * NE + kbase + kk] = f32_to_bf16(src[(size_t)(kbase + kk) * NE + n]);
    }
}

// ---------------------------------------------------------------------------
// Kernel 3: QKV projection GEMM (wave = 16M x 64N tile, bf16 WMMA)
// grid = (NE/64, NROWS/64, 3)   z: 0=Q 1=K 2=V ; out in [b,h,s,d] bf16
// ---------------------------------------------------------------------------
__global__ __launch_bounds__(128)
void qkv_gemm_kernel(const u16* __restrict__ xb,
                     const u16* __restrict__ Wqt,
                     const u16* __restrict__ Wkt,
                     const u16* __restrict__ Wvt,
                     u16* __restrict__ qo,
                     u16* __restrict__ ko,
                     u16* __restrict__ vo) {
    const u16* Wt;
    u16* dst;
    if (blockIdx.z == 0)      { Wt = Wqt; dst = qo; }
    else if (blockIdx.z == 1) { Wt = Wkt; dst = ko; }
    else                      { Wt = Wvt; dst = vo; }

    const int lane  = threadIdx.x & 31;
    const int wave  = threadIdx.x >> 5;
    const int mlane = lane & 15;
    const int half  = lane >> 4;
    const int m0    = blockIdx.y * 64 + wave * 16;
    const int n0    = blockIdx.x * 64;

    const v8f vzero = {0.f,0.f,0.f,0.f,0.f,0.f,0.f,0.f};
    v8f acc[4] = {vzero, vzero, vzero, vzero};

    const u16* arow = xb + (size_t)(m0 + mlane) * NE;

    for (int k0 = 0; k0 < NE; k0 += 32) {
        ABFrag a;
#pragma unroll
        for (int j = 0; j < 8; ++j) {
            int kk = k0 + ((j >> 2) << 4) + (half << 3) + ((j & 3) << 1);
            a.u[j] = *(const u32*)(arow + kk);
        }
#pragma unroll
        for (int t = 0; t < 4; ++t) {
            ABFrag bfr;
            const u16* brow = Wt + (size_t)(n0 + t * 16 + mlane) * NE;
#pragma unroll
            for (int j = 0; j < 8; ++j) {
                int kk = k0 + ((j >> 2) << 4) + (half << 3) + ((j & 3) << 1);
                bfr.u[j] = *(const u32*)(brow + kk);
            }
            acc[t] = wmma_bf16(a, bfr, acc[t]);
        }
    }

#pragma unroll
    for (int t = 0; t < 4; ++t) {
#pragma unroll
        for (int r = 0; r < 8; ++r) {
            int m = m0 + r + 8 * half;
            int n = n0 + t * 16 + mlane;
            int b = m >> 11, s = m & (NS - 1);
            int h = n >> 6,  d = n & (HD - 1);
            dst[(((size_t)(b * NH + h)) * NS + s) * HD + d] = f32_to_bf16(acc[t][r]);
        }
    }
}

// ---------------------------------------------------------------------------
// Attention tile body: S = Q*K^T, online softmax (-50 causal penalty), O += P*V
// ---------------------------------------------------------------------------
__device__ __forceinline__ void attn_tile(const u16* __restrict__ ldsK,
                                          const u16* __restrict__ ldsV,
                                          u16* __restrict__ Pw,
                                          const ABFrag* qa, v8f* o,
                                          float* mrow, float* lrow,
                                          int k0, int q0, int mlane, int half) {
    const v8f vzero = {0.f,0.f,0.f,0.f,0.f,0.f,0.f,0.f};
    v8f sc[2] = {vzero, vzero};
#pragma unroll
    for (int f = 0; f < 2; ++f) {
#pragma unroll
        for (int ds = 0; ds < 2; ++ds) {
            ABFrag bk;   // B frag: reduction over d, N = k-position
            const u16* krow = ldsK + (f * 16 + mlane) * HD;
#pragma unroll
            for (int j = 0; j < 8; ++j) {
                int dd = ds * 32 + ((j >> 2) << 4) + (half << 3) + ((j & 3) << 1);
                bk.u[j] = *(const u32*)(krow + dd);
            }
            sc[f] = wmma_bf16(qa[ds], bk, sc[f]);
        }
    }

    float tmax[8];
#pragma unroll
    for (int r = 0; r < 8; ++r) tmax[r] = -1e30f;
#pragma unroll
    for (int f = 0; f < 2; ++f) {
#pragma unroll
        for (int r = 0; r < 8; ++r) {
            float val = sc[f][r] * SCALE;
            int kj = k0 + f * 16 + mlane;
            int qi = q0 + r + 8 * half;
            if (kj > qi) val += MASKPEN;
            sc[f][r] = val;
            tmax[r] = fmaxf(tmax[r], val);
        }
    }
#pragma unroll
    for (int off = 1; off < 16; off <<= 1)
#pragma unroll
        for (int r = 0; r < 8; ++r)
            tmax[r] = fmaxf(tmax[r], __shfl_xor(tmax[r], off, 32));

    float alpha[8];
#pragma unroll
    for (int r = 0; r < 8; ++r) {
        float mn = fmaxf(mrow[r], tmax[r]);
        alpha[r] = __expf(mrow[r] - mn);
        mrow[r] = mn;
    }
    float rs[8];
#pragma unroll
    for (int r = 0; r < 8; ++r) rs[r] = 0.f;
#pragma unroll
    for (int f = 0; f < 2; ++f) {
#pragma unroll
        for (int r = 0; r < 8; ++r) {
            float p = __expf(sc[f][r] - mrow[r]);
            sc[f][r] = p;
            rs[r] += p;
        }
    }
#pragma unroll
    for (int off = 1; off < 16; off <<= 1)
#pragma unroll
        for (int r = 0; r < 8; ++r)
            rs[r] += __shfl_xor(rs[r], off, 32);
#pragma unroll
    for (int r = 0; r < 8; ++r) lrow[r] = lrow[r] * alpha[r] + rs[r];
#pragma unroll
    for (int t = 0; t < 4; ++t)
#pragma unroll
        for (int r = 0; r < 8; ++r) o[t][r] *= alpha[r];

    // P (C layout) -> wave-private LDS -> A fragment (in-order within a wave)
#pragma unroll
    for (int f = 0; f < 2; ++f)
#pragma unroll
        for (int r = 0; r < 8; ++r)
            Pw[(r + 8 * half) * 32 + f * 16 + mlane] = f32_to_bf16(sc[f][r]);
    ABFrag pa;
#pragma unroll
    for (int j = 0; j < 8; ++j) {
        int kk = ((j >> 2) << 4) + (half << 3) + ((j & 3) << 1);
        pa.u[j] = *(const u32*)(Pw + mlane * 32 + kk);
    }

#pragma unroll
    for (int t = 0; t < 4; ++t) {
        ABFrag vf;   // B frag: reduction over kpos, N = d
        int dd = t * 16 + mlane;
#pragma unroll
        for (int j = 0; j < 8; ++j) {
            int kp = ((j >> 2) << 4) + (half << 3) + ((j & 3) << 1);
            vf.s[2 * j]     = ldsV[kp * HD + dd];
            vf.s[2 * j + 1] = ldsV[(kp + 1) * HD + dd];
        }
        o[t] = wmma_bf16(pa, vf, o[t]);
    }
}

// ---------------------------------------------------------------------------
// Kernel 4: flash-style attention.  q,k,v: bf16 [b,h,s,d].
//   ctx out: bf16 [b, s, h*64+d] (row-major [8192,1024])
// block = 128 threads (4 waves), each wave owns a 16-row Q tile.
// Dynamic LDS layout (u16 elems):
//   [0,4096)     K ping-pong  2 x (32x64)
//   [4096,8192)  V ping-pong  2 x (32x64)
//   [8192,10240) P slabs      4 waves x (16x32)
// => 20480 bytes dynamic shared.
// ---------------------------------------------------------------------------
__global__ __launch_bounds__(128)
void attn_kernel(const u16* __restrict__ q,
                 const u16* __restrict__ k,
                 const u16* __restrict__ v,
                 u16* __restrict__ ctx) {
    extern __shared__ u16 smem[];
    u16* ldsK = smem;              // 2 buffers of 2048 elems
    u16* ldsV = smem + 4096;       // 2 buffers of 2048 elems
    u16* ldsP = smem + 8192;       // 4 wave slabs of 512 elems

    const int bh    = blockIdx.x;
    const int b     = bh >> 4;
    const int h     = bh & (NH - 1);
    const int lane  = threadIdx.x & 31;
    const int wave  = threadIdx.x >> 5;
    const int mlane = lane & 15;
    const int half  = lane >> 4;
    const int q0    = blockIdx.y * 64 + wave * 16;

    const u16* Qh = q + (size_t)bh * NS * HD;
    const u16* Kh = k + (size_t)bh * NS * HD;
    const u16* Vh = v + (size_t)bh * NS * HD;

    // Q A-fragments for d = [0,32) and [32,64)
    ABFrag qa[2];
    const u16* qrow = Qh + (size_t)(q0 + mlane) * HD;
#pragma unroll
    for (int ds = 0; ds < 2; ++ds) {
#pragma unroll
        for (int j = 0; j < 8; ++j) {
            int dd = ds * 32 + ((j >> 2) << 4) + (half << 3) + ((j & 3) << 1);
            qa[ds].u[j] = *(const u32*)(qrow + dd);
        }
    }

    const v8f vzero = {0.f,0.f,0.f,0.f,0.f,0.f,0.f,0.f};
    v8f o[4] = {vzero, vzero, vzero, vzero};
    float mrow[8], lrow[8];
#pragma unroll
    for (int r = 0; r < 8; ++r) { mrow[r] = -1e30f; lrow[r] = 0.f; }

    u16* Pw = ldsP + wave * 512;
    const int NT = NS / 32;

#if HAVE_TDM
    // ---- Tensor Data Mover double-buffered pipeline -----------------------
    if (wave == 0) {
        // prologue: tile 0 -> buffer 0 (K then V; in-order completion)
        tdm_load_tile_2d((unsigned long long)(size_t)Kh, 0u,
                         32u, HD, HD, NS, HD);
        tdm_load_tile_2d((unsigned long long)(size_t)Vh, 8192u,
                         32u, HD, HD, NS, HD);
    }
    for (int i = 0; i < NT; ++i) {
        if (wave == 0) {
            if (i + 1 < NT) {
                unsigned buf = (unsigned)((i + 1) & 1);
                tdm_load_tile_2d(
                    (unsigned long long)(size_t)(Kh + (size_t)(i + 1) * 32 * HD),
                    buf * 4096u, 32u, HD, HD, (unsigned)(NS - (i + 1) * 32), HD);
                tdm_load_tile_2d(
                    (unsigned long long)(size_t)(Vh + (size_t)(i + 1) * 32 * HD),
                    8192u + buf * 4096u, 32u, HD, HD,
                    (unsigned)(NS - (i + 1) * 32), HD);
                __builtin_amdgcn_s_wait_tensorcnt(2);  // tile i landed
            } else {
                __builtin_amdgcn_s_wait_tensorcnt(0);  // last tile landed
            }
        }
        __syncthreads();
        attn_tile(ldsK + (size_t)(i & 1) * 2048, ldsV + (size_t)(i & 1) * 2048,
                  Pw, qa, o, mrow, lrow, i * 32, q0, mlane, half);
        __syncthreads();  // compute done before this buffer's next TDM write
    }
#else
    // ---- fallback: synchronous cooperative staging ------------------------
    for (int i = 0; i < NT; ++i) {
        {
            const uint4* gk = (const uint4*)(Kh + (size_t)i * 32 * HD);
            const uint4* gv = (const uint4*)(Vh + (size_t)i * 32 * HD);
            uint4* sk = (uint4*)ldsK;
            uint4* sv = (uint4*)ldsV;
            int t = threadIdx.x;
            sk[t] = gk[t]; sk[t + 128] = gk[t + 128];
            sv[t] = gv[t]; sv[t + 128] = gv[t + 128];
        }
        if (i + 1 < NT)
            __builtin_prefetch(Kh + (size_t)(i + 1) * 32 * HD, 0, 3);
        __syncthreads();
        attn_tile(ldsK, ldsV, Pw, qa, o, mrow, lrow, i * 32, q0, mlane, half);
        __syncthreads();
    }
#endif

    // epilogue: normalize, write ctx [b, s, h*64+d]
#pragma unroll
    for (int t = 0; t < 4; ++t) {
#pragma unroll
        for (int r = 0; r < 8; ++r) {
            float val = o[t][r] / lrow[r];
            int s = q0 + r + 8 * half;
            int d = t * 16 + mlane;
            ctx[((size_t)(b * NS + s)) * NE + h * HD + d] = f32_to_bf16(val);
        }
    }
}

// ---------------------------------------------------------------------------
// Kernel 5: output GEMM  out(f32) = ctx(bf16 [8192x1024]) @ Wr  (Wrt = Wr^T)
// ---------------------------------------------------------------------------
__global__ __launch_bounds__(128)
void out_gemm_kernel(const u16* __restrict__ ctx,
                     const u16* __restrict__ Wrt,
                     float* __restrict__ out) {
    const int lane  = threadIdx.x & 31;
    const int wave  = threadIdx.x >> 5;
    const int mlane = lane & 15;
    const int half  = lane >> 4;
    const int m0    = blockIdx.y * 64 + wave * 16;
    const int n0    = blockIdx.x * 64;

    const v8f vzero = {0.f,0.f,0.f,0.f,0.f,0.f,0.f,0.f};
    v8f acc[4] = {vzero, vzero, vzero, vzero};

    const u16* arow = ctx + (size_t)(m0 + mlane) * NE;
    for (int k0 = 0; k0 < NE; k0 += 32) {
        ABFrag a;
#pragma unroll
        for (int j = 0; j < 8; ++j) {
            int kk = k0 + ((j >> 2) << 4) + (half << 3) + ((j & 3) << 1);
            a.u[j] = *(const u32*)(arow + kk);
        }
#pragma unroll
        for (int t = 0; t < 4; ++t) {
            ABFrag bfr;
            const u16* brow = Wrt + (size_t)(n0 + t * 16 + mlane) * NE;
#pragma unroll
            for (int j = 0; j < 8; ++j) {
                int kk = k0 + ((j >> 2) << 4) + (half << 3) + ((j & 3) << 1);
                bfr.u[j] = *(const u32*)(brow + kk);
            }
            acc[t] = wmma_bf16(a, bfr, acc[t]);
        }
    }
#pragma unroll
    for (int t = 0; t < 4; ++t)
#pragma unroll
        for (int r = 0; r < 8; ++r) {
            int m = m0 + r + 8 * half;
            int n = n0 + t * 16 + mlane;
            out[(size_t)m * NE + n] = acc[t][r];
        }
}

// ---------------------------------------------------------------------------
// Launcher
// ---------------------------------------------------------------------------
extern "C" void kernel_launch(void* const* d_in, const int* in_sizes, int n_in,
                              void* d_out, int out_size, void* d_ws, size_t ws_size,
                              hipStream_t stream) {
    (void)in_sizes; (void)n_in; (void)out_size; (void)ws_size;
    const float* x  = (const float*)d_in[0];
    const float* Wq = (const float*)d_in[1];
    const float* Wk = (const float*)d_in[2];
    const float* Wv = (const float*)d_in[3];
    const float* Wr = (const float*)d_in[4];
    float* out = (float*)d_out;

    // Workspace layout: 88 MB total (fits global L2)
    char* p = (char*)d_ws;
    u16* xb  = (u16*)p; p += (size_t)NROWS * NE * 2;   // 16 MB
    u16* Wqt = (u16*)p; p += (size_t)NE * NE * 2;      //  2 MB
    u16* Wkt = (u16*)p; p += (size_t)NE * NE * 2;
    u16* Wvt = (u16*)p; p += (size_t)NE * NE * 2;
    u16* Wrt = (u16*)p; p += (size_t)NE * NE * 2;
    u16* qb  = (u16*)p; p += (size_t)NROWS * NE * 2;   // [b,h,s,d]
    u16* kb  = (u16*)p; p += (size_t)NROWS * NE * 2;
    u16* vb  = (u16*)p; p += (size_t)NROWS * NE * 2;
    u16* ctx = (u16*)p; p += (size_t)NROWS * NE * 2;

    const int nX = NROWS * NE;
    cvt_bf16_kernel<<<(nX + 255) / 256, 256, 0, stream>>>(x, xb, nX);
    cvt_transpose_kernel<<<dim3(32, 32), 32, 0, stream>>>(Wq, Wqt);
    cvt_transpose_kernel<<<dim3(32, 32), 32, 0, stream>>>(Wk, Wkt);
    cvt_transpose_kernel<<<dim3(32, 32), 32, 0, stream>>>(Wv, Wvt);
    cvt_transpose_kernel<<<dim3(32, 32), 32, 0, stream>>>(Wr, Wrt);

    qkv_gemm_kernel<<<dim3(NE / 64, NROWS / 64, 3), 128, 0, stream>>>(
        xb, Wqt, Wkt, Wvt, qb, kb, vb);

    attn_kernel<<<dim3(NB * NH, NS / 64), 128, 20480, stream>>>(qb, kb, vb, ctx);

    out_gemm_kernel<<<dim3(NE / 64, NROWS / 64), 128, 0, stream>>>(ctx, Wrt, out);
}
